// SearchRNN_49787260895857
// MI455X (gfx1250) — compile-verified, matrix operand order checked by reading, not run on hardware
//
#include <hip/hip_runtime.h>
#include <hip/hip_bf16.h>
#include <math.h>

// ---------------------------------------------------------------------------
// Types for CDNA5 WMMA (wave32): v_wmma_f32_16x16x32_bf16
// ---------------------------------------------------------------------------
typedef __attribute__((ext_vector_type(16))) __bf16 v16bf;
typedef __attribute__((ext_vector_type(8)))  float  v8f;

#define B_   32
#define S_   64
#define T_   64
#define E_   512
#define H_   512
#define G3_  1536      // 3*H
#define DH_  1024      // 2*Hs (bidirectional hidden)
#define XD_  1536      // Et + 2*Hs (decoder input)
#define V_   32000
#define NEGV (-1e9f)

// ---------------------------------------------------------------------------
// WMMA fragment helpers
// A fragment 16x32 (MxK) bf16 from row-major [rows, ld]:
//   lane 0-15  (m = lane)   : K = k0+0..7   and k0+16..23
//   lane 16-31 (m = lane-16): K = k0+8..15  and k0+24..31
// B operand for C = A * W^T loads W[N][K] row-major with the same pattern
// (lane selects output column n = row of W).
// ---------------------------------------------------------------------------
__device__ __forceinline__ v16bf load_frag16(const __bf16* __restrict__ p,
                                             int ld, int row0, int k0) {
  const int lane = threadIdx.x & 31;
  const int r    = row0 + (lane & 15);
  const int koff = k0 + ((lane >> 4) << 3);
  const __bf16* b0 = p + (size_t)r * (size_t)ld + koff;
  v16bf f;
#pragma unroll
  for (int i = 0; i < 8; ++i) { f[i] = b0[i]; f[i + 8] = b0[i + 16]; }
  return f;
}

// Same fragment pattern but sourced from an LDS slice laid out as
// [16 rows][64 bytes (32 bf16) of K] contiguous.
__device__ __forceinline__ v16bf load_frag_lds(const __bf16* slice, int lane) {
  const __bf16* rp = slice + (lane & 15) * 32 + ((lane >> 4) << 3);
  v16bf f;
#pragma unroll
  for (int i = 0; i < 8; ++i) { f[i] = rp[i]; f[8 + i] = rp[16 + i]; }
  return f;
}

// C/D 16x16 f32: VGPR r -> row = r + 8*(lane>=16), col = lane%16
__device__ __forceinline__ void store_tile_f32(float* __restrict__ out, int ld,
                                               int m0, int n0, v8f c) {
  const int lane  = threadIdx.x & 31;
  const int col   = n0 + (lane & 15);
  const int rbase = m0 + ((lane >> 4) << 3);
#pragma unroll
  for (int r = 0; r < 8; ++r)
    out[(size_t)(rbase + r) * (size_t)ld + col] = c[r];
}

// ---------------------------------------------------------------------------
// Async stage of one W k-slice (16 rows x 32 K bf16 = 1KB) into an LDS slab,
// via CDNA5 global->LDS async DMA (tracked by ASYNCcnt).
// Chunk c (of 64 x 16B): row = c>>2, 16B block (c&3). Lane does c=lane and
// c=lane+32 (rows 0-7 then 8-15).
// ---------------------------------------------------------------------------
__device__ __forceinline__ void stage_w_async(const __bf16* __restrict__ W,
                                              unsigned ldsslab, int n0, int k0,
                                              int lane) {
  const char* base = (const char*)W;
  const char* p1 = base + (size_t)(n0 + (lane >> 2)) * 1024
                        + (size_t)k0 * 2 + (size_t)(lane & 3) * 16;
  const char* p2 = p1 + 8 * 1024;  // rows 8..15
  unsigned l1 = ldsslab + (unsigned)lane * 16;
  unsigned l2 = l1 + 512;
  // make sure prior ds reads of this buffer have completed
  asm volatile("s_wait_dscnt 0x0" ::: "memory");
  asm volatile("global_load_async_to_lds_b128 %0, %1, off"
               :: "v"(l1), "v"(p1) : "memory");
  asm volatile("global_load_async_to_lds_b128 %0, %1, off"
               :: "v"(l2), "v"(p2) : "memory");
}

// ---------------------------------------------------------------------------
// Generic GEMM: C[M,N] = A[M,K](bf16) * W[N,K](bf16)^T + bias[N]
// grid = (N/16, M/16), one wave (32 threads) per 16x16 output tile.
// ---------------------------------------------------------------------------
__global__ void gemm_bf16_wmma(const __bf16* __restrict__ A, int lda,
                               const __bf16* __restrict__ W, int ldw,
                               const float* __restrict__ bias,
                               float* __restrict__ C, int ldc, int K) {
  const int n0 = blockIdx.x * 16;
  const int m0 = blockIdx.y * 16;
  v8f acc = {};
  for (int k0 = 0; k0 < K; k0 += 32) {
    v16bf a = load_frag16(A, lda, m0, k0);
    v16bf b = load_frag16(W, ldw, n0, k0);
    acc = __builtin_amdgcn_wmma_f32_16x16x32_bf16(false, a, false, b,
                                                  (short)0, acc, false, false);
  }
  if (bias) {
    const float bv = bias[n0 + (threadIdx.x & 15)];
#pragma unroll
    for (int r = 0; r < 8; ++r) acc[r] += bv;
  }
  store_tile_f32(C, ldc, m0, n0, acc);
}

// ---------------------------------------------------------------------------
// Utility kernels
// ---------------------------------------------------------------------------
__global__ void cvt_f32_bf16(const float* __restrict__ in,
                             __bf16* __restrict__ out, int n) {
  int i = blockIdx.x * blockDim.x + threadIdx.x;
  if (i < n) out[i] = (__bf16)in[i];
}

__global__ void zero_u32(unsigned int* __restrict__ p, int n) {
  int i = blockIdx.x * blockDim.x + threadIdx.x;
  if (i < n) p[i] = 0u;
}

// gather embedding rows -> bf16 [rows, 512]
__global__ void gather_emb(const int* __restrict__ seqs,
                           const float* __restrict__ emb,
                           __bf16* __restrict__ out, int total) {
  int i = blockIdx.x * blockDim.x + threadIdx.x;
  if (i >= total) return;
  int row = i >> 9, j = i & 511;
  out[i] = (__bf16)emb[(size_t)seqs[row] * 512 + j];
}

__device__ __forceinline__ float sigmoidf_(float x) {
  return 1.f / (1.f + expf(-x));
}

// ---------------------------------------------------------------------------
// Encoder gate update (both directions in one launch). step i:
//   forward t=i, backward t=S-1-i.
// ---------------------------------------------------------------------------
__global__ void enc_gate(const float* __restrict__ gxF, const float* __restrict__ gxB,
                         const float* __restrict__ ghF, const float* __restrict__ ghB,
                         float* __restrict__ hF, float* __restrict__ hB,
                         __bf16* __restrict__ hFb, __bf16* __restrict__ hBb,
                         float* __restrict__ hidden, const int* __restrict__ src_len,
                         int step) {
  int tid = blockIdx.x * blockDim.x + threadIdx.x;  // 2*32*512
  if (tid >= 2 * B_ * H_) return;
  int d   = tid >> 14;          // 32*512 = 16384 per dir
  int rem = tid & 16383;
  int b = rem >> 9, j = rem & 511;
  int t = d ? (S_ - 1 - step) : step;
  const float* gx = (d ? gxB : gxF) + ((size_t)(b * S_ + t)) * G3_;
  const float* gh = (d ? ghB : ghF) + (size_t)b * G3_;
  float* h   = d ? hB : hF;
  __bf16* hb = d ? hBb : hFb;
  float hp = h[b * H_ + j];
  float r = sigmoidf_(gx[j] + gh[j]);
  float z = sigmoidf_(gx[H_ + j] + gh[H_ + j]);
  float n = tanhf(gx[2 * H_ + j] + r * gh[2 * H_ + j]);
  float hnew = (1.f - z) * n + z * hp;
  bool mask = t < src_len[b];
  float hu = mask ? hnew : hp;
  h[b * H_ + j]  = hu;
  hb[b * H_ + j] = (__bf16)hu;
  hidden[((size_t)(b * S_ + t)) * DH_ + d * H_ + j] = mask ? hu : 0.f;
}

// ---------------------------------------------------------------------------
// Bahdanau attention + context. One block per batch row.
// ---------------------------------------------------------------------------
__global__ void attn_ctx(const float* __restrict__ Uh, const float* __restrict__ Wh,
                         const float* __restrict__ Av, const float* __restrict__ hidden,
                         const int* __restrict__ src_len, float* __restrict__ ctx) {
  __shared__ float e_s[S_];
  __shared__ float alpha[S_];
  __shared__ float red[2];
  const int b = blockIdx.x;
  const int wave = threadIdx.x >> 5, lane = threadIdx.x & 31;
  const int len = src_len[b];
  for (int s = wave; s < S_; s += 8) {
    float acc = 0.f;
    const float* uh = Uh + ((size_t)(b * S_ + s)) * H_;
    const float* wh = Wh + (size_t)b * H_;
    for (int h = lane; h < H_; h += 32)
      acc += tanhf(uh[h] + wh[h]) * Av[h];
#pragma unroll
    for (int off = 16; off; off >>= 1) acc += __shfl_xor(acc, off, 32);
    if (lane == 0) e_s[s] = (s < len) ? acc : NEGV;
  }
  __syncthreads();
  if (threadIdx.x == 0) {
    float M = -INFINITY;
    for (int s = 0; s < S_; ++s) M = fmaxf(M, e_s[s]);
    float Ssum = 0.f;
    for (int s = 0; s < S_; ++s) Ssum += expf(e_s[s] - M);
    red[0] = M; red[1] = Ssum;
  }
  __syncthreads();
  {
    float M = red[0], Ssum = red[1];
    for (int s = threadIdx.x; s < S_; s += blockDim.x)
      alpha[s] = expf(e_s[s] - M) / Ssum;
  }
  __syncthreads();
  for (int d = threadIdx.x; d < DH_; d += blockDim.x) {
    float acc = 0.f;
    const float* hb = hidden + (size_t)b * S_ * DH_ + d;
#pragma unroll 4
    for (int s = 0; s < S_; ++s) acc += alpha[s] * hb[(size_t)s * DH_];
    ctx[(size_t)b * DH_ + d] = acc;
  }
}

// Build decoder GRU input x = [tgt_emb[tok], ctx] (bf16)
__global__ void build_x(const int* __restrict__ tgt_seqs,
                        const float* __restrict__ tgt_emb,
                        const float* __restrict__ ctx,
                        __bf16* __restrict__ xcat, int t) {
  int tid = blockIdx.x * blockDim.x + threadIdx.x;  // 32*1536
  if (tid >= B_ * XD_) return;
  int b = tid / XD_, j = tid % XD_;
  float v;
  if (j < E_) {
    int tok = tgt_seqs[b * T_ + t];
    v = tgt_emb[(size_t)tok * E_ + j];
  } else {
    v = ctx[(size_t)b * DH_ + (j - E_)];
  }
  xcat[tid] = (__bf16)v;
}

// Decoder gate update, writes dec_out row t
__global__ void dec_gate(const float* __restrict__ gx, const float* __restrict__ gh,
                         float* __restrict__ h, __bf16* __restrict__ hb,
                         float* __restrict__ dec_out,
                         const int* __restrict__ tgt_len, int t) {
  int tid = blockIdx.x * blockDim.x + threadIdx.x;  // 32*512
  if (tid >= B_ * H_) return;
  int b = tid >> 9, j = tid & 511;
  const float* gxr = gx + (size_t)b * G3_;
  const float* ghr = gh + (size_t)b * G3_;
  float hp = h[tid];
  float r = sigmoidf_(gxr[j] + ghr[j]);
  float z = sigmoidf_(gxr[H_ + j] + ghr[H_ + j]);
  float n = tanhf(gxr[2 * H_ + j] + r * ghr[2 * H_ + j]);
  float hnew = (1.f - z) * n + z * hp;
  bool mask = t < tgt_len[b];
  float hu = mask ? hnew : hp;
  h[tid]  = hu;
  hb[tid] = (__bf16)hu;
  dec_out[((size_t)(b * T_ + t)) * H_ + j] = mask ? hnew : 0.f;
}

// ---------------------------------------------------------------------------
// Fused vocab projection + online log-softmax stats + picked-logit capture.
// grid = 128 blocks (16 rows each), 8 waves/block.
//  - A tile (16x512) is hoisted into registers (16 WMMA fragments, 128 VGPRs)
//  - W k-slices stream global->LDS via async DMA (ASYNCcnt), double-buffered
//    per wave (2 x 1KB slabs), consumed through ds loads into WMMA fragments.
//  - Per-lane online (max, sumexp) over the column subset each lane sees;
//    merged across lanes/waves in LDS at the end.
// ---------------------------------------------------------------------------
__global__ void loss_tiles(const __bf16* __restrict__ Abf,   // [2048,512]
                           const __bf16* __restrict__ Wbf,   // [32000,512]
                           const float* __restrict__ outb,   // [32000]
                           const int* __restrict__ tgt_seqs,
                           float* __restrict__ picked,       // [2048]
                           float* __restrict__ lse) {        // [2048]
  __shared__ __align__(16) __bf16 shW[8][2][512];   // 8 waves x double buffer
  __shared__ float smx[256][8];
  __shared__ float sse[256][8];
  const int m0   = blockIdx.x * 16;
  const int wave = threadIdx.x >> 5;
  const int lane = threadIdx.x & 31;
  const int rowBase = m0 + ((lane >> 4) << 3);

  // LDS byte offset of this wave's staging slab (flat addr low 32 bits = DS addr)
  const unsigned ldsbase = (unsigned)(size_t)(&shW[wave][0][0]);

  // Hoist the A tile: 16 K-step fragments held in registers for all 250 tiles
  v16bf afr[16];
#pragma unroll
  for (int k = 0; k < 16; ++k) afr[k] = load_frag16(Abf, 512, m0, k << 5);

  float mx[8], se[8];
  int goals[8];
#pragma unroll
  for (int r = 0; r < 8; ++r) {
    mx[r] = -INFINITY; se[r] = 0.f;
    int gm = rowBase + r;
    int b = gm / T_, t = gm % T_;
    goals[r] = (t < T_ - 1) ? tgt_seqs[b * T_ + t + 1] : 0;
  }

  const int NT = V_ / 16;  // 2000
  for (int tile = wave; tile < NT; tile += 8) {
    const int n0 = tile * 16;
    if (tile + 8 < NT)   // L2 prefetch of next tile's W block
      __builtin_prefetch((const char*)Wbf + (size_t)(tile + 8) * 16 * 1024, 0, 1);

    v8f acc = {};
    stage_w_async(Wbf, ldsbase, n0, 0, lane);       // prologue: slice 0 -> buf0
#pragma unroll
    for (int ki = 0; ki < 16; ++ki) {
      const int cur = ki & 1;
      if (ki < 15) {
        stage_w_async(Wbf, ldsbase + (unsigned)((cur ^ 1) * 1024),
                      n0, (ki + 1) << 5, lane);
        asm volatile("s_wait_asynccnt 0x2" ::: "memory");  // prev slice landed
      } else {
        asm volatile("s_wait_asynccnt 0x0" ::: "memory");  // final slice landed
      }
      v16bf w = load_frag_lds(&shW[wave][cur][0], lane);
      acc = __builtin_amdgcn_wmma_f32_16x16x32_bf16(false, afr[ki], false, w,
                                                    (short)0, acc, false, false);
    }

    const int col = n0 + (lane & 15);
    const float bv = outb[col];
#pragma unroll
    for (int r = 0; r < 8; ++r) {
      float v = acc[r] + bv;
      if (col == goals[r]) picked[rowBase + r] = v;
      if (v > mx[r]) { se[r] = se[r] * expf(mx[r] - v) + 1.f; mx[r] = v; }
      else           { se[r] += expf(v - mx[r]); }
    }
  }

#pragma unroll
  for (int r = 0; r < 8; ++r) { smx[threadIdx.x][r] = mx[r]; sse[threadIdx.x][r] = se[r]; }
  __syncthreads();

  if (threadIdx.x < 16) {
    const int m = threadIdx.x;
    const int rsel  = (m < 8) ? m : m - 8;
    const int lbase = (m < 8) ? 0 : 16;
    float M = -INFINITY;
    for (int w = 0; w < 8; ++w)
      for (int l = 0; l < 16; ++l)
        M = fmaxf(M, smx[w * 32 + lbase + l][rsel]);
    float Ssum = 0.f;
    for (int w = 0; w < 8; ++w)
      for (int l = 0; l < 16; ++l) {
        int idx = w * 32 + lbase + l;
        Ssum += sse[idx][rsel] * expf(smx[idx][rsel] - M);
      }
    lse[m0 + m] = M + logf(Ssum);
  }
}

// Final masked NLL reduction -> scalar loss
__global__ void final_loss(const float* __restrict__ picked,
                           const float* __restrict__ lse,
                           const int* __restrict__ tgt_seqs,
                           float* __restrict__ out) {
  __shared__ float sn[256], sd[256];
  float num = 0.f, den = 0.f;
  for (int m = threadIdx.x; m < B_ * T_; m += 256) {
    int b = m >> 6, t = m & 63;
    int goal = (t < T_ - 1) ? tgt_seqs[b * T_ + t + 1] : 0;
    if (goal != 0) { num += picked[m] - lse[m]; den += 1.f; }
  }
  sn[threadIdx.x] = num; sd[threadIdx.x] = den;
  __syncthreads();
  for (int s = 128; s; s >>= 1) {
    if (threadIdx.x < s) { sn[threadIdx.x] += sn[threadIdx.x + s]; sd[threadIdx.x] += sd[threadIdx.x + s]; }
    __syncthreads();
  }
  if (threadIdx.x == 0) out[0] = -sn[0] / sd[0];
}

// ---------------------------------------------------------------------------
// Host side
// ---------------------------------------------------------------------------
extern "C" void kernel_launch(void* const* d_in, const int* in_sizes, int n_in,
                              void* d_out, int out_size, void* d_ws, size_t ws_size,
                              hipStream_t stream) {
  (void)in_sizes; (void)n_in; (void)out_size; (void)ws_size;
  const int*   src_seqs  = (const int*)d_in[0];
  const int*   src_len   = (const int*)d_in[1];
  const int*   tgt_seqs  = (const int*)d_in[2];
  const int*   tgt_len   = (const int*)d_in[3];
  const float* src_emb   = (const float*)d_in[4];
  const float* Wih_f     = (const float*)d_in[5];
  const float* Whh_f     = (const float*)d_in[6];
  const float* bih_f     = (const float*)d_in[7];
  const float* bhh_f     = (const float*)d_in[8];
  const float* Wih_b     = (const float*)d_in[9];
  const float* Whh_b     = (const float*)d_in[10];
  const float* bih_b     = (const float*)d_in[11];
  const float* bhh_b     = (const float*)d_in[12];
  const float* tgt_emb   = (const float*)d_in[13];
  const float* dWih      = (const float*)d_in[14];
  const float* dWhh      = (const float*)d_in[15];
  const float* dbih      = (const float*)d_in[16];
  const float* dbhh      = (const float*)d_in[17];
  const float* U_w       = (const float*)d_in[18];
  const float* U_b       = (const float*)d_in[19];
  const float* A_W       = (const float*)d_in[20];
  const float* A_b       = (const float*)d_in[21];
  const float* A_v       = (const float*)d_in[22];
  const float* out_w     = (const float*)d_in[23];
  const float* out_b     = (const float*)d_in[24];

  // workspace carve-out (256B aligned regions)
  char* cur = (char*)d_ws;
  auto alloc = [&](size_t bytes) -> void* {
    void* p = (void*)cur;
    cur += (bytes + 255) & ~(size_t)255;
    return p;
  };
  __bf16* wWihF  = (__bf16*)alloc((size_t)G3_ * E_ * 2);
  __bf16* wWihB  = (__bf16*)alloc((size_t)G3_ * E_ * 2);
  __bf16* wWhhF  = (__bf16*)alloc((size_t)G3_ * H_ * 2);
  __bf16* wWhhB  = (__bf16*)alloc((size_t)G3_ * H_ * 2);
  __bf16* wUw    = (__bf16*)alloc((size_t)H_ * DH_ * 2);
  __bf16* wDWih  = (__bf16*)alloc((size_t)G3_ * XD_ * 2);
  __bf16* wDWhh  = (__bf16*)alloc((size_t)G3_ * H_ * 2);
  __bf16* wAW    = (__bf16*)alloc((size_t)H_ * H_ * 2);
  __bf16* wOutW  = (__bf16*)alloc((size_t)V_ * H_ * 2);
  __bf16* eSrc   = (__bf16*)alloc((size_t)B_ * S_ * E_ * 2);
  float*  gxF    = (float*) alloc((size_t)B_ * S_ * G3_ * 4);
  float*  gxB    = (float*) alloc((size_t)B_ * S_ * G3_ * 4);
  float*  ghF    = (float*) alloc((size_t)B_ * G3_ * 4);
  float*  ghB    = (float*) alloc((size_t)B_ * G3_ * 4);
  float*  hF     = (float*) alloc((size_t)B_ * H_ * 4);
  float*  hB     = (float*) alloc((size_t)B_ * H_ * 4);
  __bf16* hFb    = (__bf16*)alloc((size_t)B_ * H_ * 2);
  __bf16* hBb    = (__bf16*)alloc((size_t)B_ * H_ * 2);
  float*  hidden = (float*) alloc((size_t)B_ * S_ * DH_ * 4);
  __bf16* hiddenb= (__bf16*)alloc((size_t)B_ * S_ * DH_ * 2);
  float*  Uh     = (float*) alloc((size_t)B_ * S_ * H_ * 4);
  float*  Wh     = (float*) alloc((size_t)B_ * H_ * 4);
  float*  ctx    = (float*) alloc((size_t)B_ * DH_ * 4);
  __bf16* xcat   = (__bf16*)alloc((size_t)B_ * XD_ * 2);
  float*  gxd    = (float*) alloc((size_t)B_ * G3_ * 4);
  float*  ghd    = (float*) alloc((size_t)B_ * G3_ * 4);
  float*  hDec   = (float*) alloc((size_t)B_ * H_ * 4);
  __bf16* hDecb  = (__bf16*)alloc((size_t)B_ * H_ * 2);
  float*  decOut = (float*) alloc((size_t)B_ * T_ * H_ * 4);
  __bf16* decOutb= (__bf16*)alloc((size_t)B_ * T_ * H_ * 2);
  float*  picked = (float*) alloc((size_t)B_ * T_ * 4);
  float*  lseArr = (float*) alloc((size_t)B_ * T_ * 4);

  auto cvt = [&](const float* src, __bf16* dst, int n) {
    cvt_f32_bf16<<<(n + 255) / 256, 256, 0, stream>>>(src, dst, n);
  };

  // ---- weight conversions (fp32 -> bf16) ----
  cvt(Wih_f, wWihF, G3_ * E_);
  cvt(Wih_b, wWihB, G3_ * E_);
  cvt(Whh_f, wWhhF, G3_ * H_);
  cvt(Whh_b, wWhhB, G3_ * H_);
  cvt(U_w,   wUw,   H_ * DH_);
  cvt(dWih,  wDWih, G3_ * XD_);
  cvt(dWhh,  wDWhh, G3_ * H_);
  cvt(A_W,   wAW,   H_ * H_);
  cvt(out_w, wOutW, V_ * H_);

  // ---- zero-init recurrent states ----
  zero_u32<<<(B_ * H_ + 255) / 256, 256, 0, stream>>>((unsigned*)hF,   B_ * H_);
  zero_u32<<<(B_ * H_ + 255) / 256, 256, 0, stream>>>((unsigned*)hB,   B_ * H_);
  zero_u32<<<(B_ * H_ + 255) / 256, 256, 0, stream>>>((unsigned*)hDec, B_ * H_);
  zero_u32<<<(B_ * H_ / 2 + 255) / 256, 256, 0, stream>>>((unsigned*)hFb,   B_ * H_ / 2);
  zero_u32<<<(B_ * H_ / 2 + 255) / 256, 256, 0, stream>>>((unsigned*)hBb,   B_ * H_ / 2);
  zero_u32<<<(B_ * H_ / 2 + 255) / 256, 256, 0, stream>>>((unsigned*)hDecb, B_ * H_ / 2);

  // ---- encoder: gather embeddings, precompute gx for all timesteps ----
  gather_emb<<<(B_ * S_ * E_ + 255) / 256, 256, 0, stream>>>(src_seqs, src_emb, eSrc,
                                                             B_ * S_ * E_);
  gemm_bf16_wmma<<<dim3(G3_ / 16, B_ * S_ / 16), 32, 0, stream>>>(
      eSrc, E_, wWihF, E_, bih_f, gxF, G3_, E_);
  gemm_bf16_wmma<<<dim3(G3_ / 16, B_ * S_ / 16), 32, 0, stream>>>(
      eSrc, E_, wWihB, E_, bih_b, gxB, G3_, E_);

  // ---- encoder recurrence (64 sequential steps, fwd+bwd) ----
  for (int step = 0; step < S_; ++step) {
    gemm_bf16_wmma<<<dim3(G3_ / 16, B_ / 16), 32, 0, stream>>>(
        hFb, H_, wWhhF, H_, bhh_f, ghF, G3_, H_);
    gemm_bf16_wmma<<<dim3(G3_ / 16, B_ / 16), 32, 0, stream>>>(
        hBb, H_, wWhhB, H_, bhh_b, ghB, G3_, H_);
    enc_gate<<<(2 * B_ * H_ + 255) / 256, 256, 0, stream>>>(
        gxF, gxB, ghF, ghB, hF, hB, hFb, hBb, hidden, src_len, step);
  }

  // ---- attention precompute: Uh = hidden @ U_w^T + U_b ----
  cvt_f32_bf16<<<(B_ * S_ * DH_ + 255) / 256, 256, 0, stream>>>(hidden, hiddenb,
                                                                B_ * S_ * DH_);
  gemm_bf16_wmma<<<dim3(H_ / 16, B_ * S_ / 16), 32, 0, stream>>>(
      hiddenb, DH_, wUw, DH_, U_b, Uh, H_, DH_);

  // ---- decoder recurrence (64 sequential steps) ----
  for (int t = 0; t < T_; ++t) {
    gemm_bf16_wmma<<<dim3(H_ / 16, B_ / 16), 32, 0, stream>>>(
        hDecb, H_, wAW, H_, A_b, Wh, H_, H_);
    attn_ctx<<<B_, 256, 0, stream>>>(Uh, Wh, A_v, hidden, src_len, ctx);
    build_x<<<(B_ * XD_ + 255) / 256, 256, 0, stream>>>(tgt_seqs, tgt_emb, ctx, xcat, t);
    gemm_bf16_wmma<<<dim3(G3_ / 16, B_ / 16), 32, 0, stream>>>(
        xcat, XD_, wDWih, XD_, dbih, gxd, G3_, XD_);
    gemm_bf16_wmma<<<dim3(G3_ / 16, B_ / 16), 32, 0, stream>>>(
        hDecb, H_, wDWhh, H_, dbhh, ghd, G3_, H_);
    dec_gate<<<(B_ * H_ + 255) / 256, 256, 0, stream>>>(
        gxd, ghd, hDec, hDecb, decOut, tgt_len, t);
  }

  // ---- fused vocab projection + online log-softmax + NLL ----
  cvt_f32_bf16<<<(B_ * T_ * H_ + 255) / 256, 256, 0, stream>>>(decOut, decOutb,
                                                               B_ * T_ * H_);
  loss_tiles<<<B_ * T_ / 16, 256, 0, stream>>>(decOutb, wOutW, out_b, tgt_seqs,
                                               picked, lseArr);
  final_loss<<<1, 256, 0, stream>>>(picked, lseArr, tgt_seqs, (float*)d_out);
}
